// TransformerEncoder_75548474736800
// MI455X (gfx1250) — matrix operand order repaired and checked by reading, not compiled, not run on hardware
//
#include <hip/hip_runtime.h>
#include <hip/hip_bf16.h>
#include <math.h>

typedef unsigned short u16;
typedef unsigned int   u32;
typedef __attribute__((ext_vector_type(16))) __bf16 v16bf;
typedef __attribute__((ext_vector_type(8)))  float  v8f;

#define DIM     512
#define TOK     512
#define BATCH   32
#define NHEAD   8
#define DHD     64
#define NHASH   8
#define BUCKET  16
#define NBUK    32
#define NCHUNK  256           // chunks per (b,h) across all hash rounds
#define BHN     (BATCH*NHEAD) // 256
#define MROWS   (BATCH*TOK)   // 16384
#define FFD     2048

// GEMM block tile
#define BM 64
#define BN 64
#define BK 64
#define LDSW 72               // u16 per LDS tile row (144B: 128B data + 16B pad)
#define TILE_U16 (BM * LDSW)  // 4608 u16 = 9216 B per tile buffer

__device__ __forceinline__ u16 f2bf(float f) {
  u32 x = __float_as_uint(f);
  x += 0x7fffu + ((x >> 16) & 1u);   // round-to-nearest-even
  return (u16)(x >> 16);
}

// gfx1250 async global->LDS copy (ASYNCcnt-tracked); lds_off = workgroup-relative
// LDS byte offset (low 32 bits of the generic pointer per ISA aperture mapping)
__device__ __forceinline__ void async_b128(unsigned lds_off, const void* gaddr) {
  asm volatile("global_load_async_to_lds_b128 %0, %1, off"
               :: "v"(lds_off), "v"(gaddr) : "memory");
}
__device__ __forceinline__ void wait_async0() {
  asm volatile("s_wait_asynccnt 0x0" ::: "memory");
}

// ---------------------------------------------------------------- organize
__global__ void k_organize(const float* __restrict__ sub, const float* __restrict__ obj,
                           float* __restrict__ X1, float* __restrict__ X2) {
  long idx = (long)blockIdx.x * blockDim.x + threadIdx.x;
  if (idx >= (long)BATCH * TOK * DIM) return;
  int c = (int)(idx % DIM);
  int t = (int)((idx / DIM) % TOK);
  int b = (int)(idx / ((long)DIM * TOK));
  float val;
  if (c == DIM - 1) {
    val = 1.0f;
  } else {
    const float* src = (t < 256) ? sub : obj;
    int p = t & 255;
    int i = p >> 4, j = p & 15;
    long base = (((long)b * 511 + c) * 32 + 2 * i) * 32 + 2 * j;
    float mx = fmaxf(fmaxf(src[base], src[base + 1]),
                     fmaxf(src[base + 32], src[base + 33]));
    int k = c >> 1;
    float freq = expf((float)(2 * k) * (-9.2103403719761836f / 511.0f));
    float ang  = (float)(i + j) * freq;
    float pe   = (c & 1) ? cosf(ang) : sinf(ang);
    val = mx + pe;
  }
  X1[idx] = val;
  X2[idx] = val;
}

// ---------------------------------------------------------------- f32 -> bf16
__global__ void k_convert(const float* __restrict__ in, u16* __restrict__ out, long n) {
  long i = (long)blockIdx.x * blockDim.x + threadIdx.x;
  if (i < n) out[i] = f2bf(in[i]);
}

// ---------------------------------------------------------------- LayerNorm -> bf16
__global__ void k_ln(const float* __restrict__ X, const float* __restrict__ g,
                     const float* __restrict__ bt, u16* __restrict__ out) {
  int row = blockIdx.x;
  int tid = threadIdx.x;                     // 256 threads, 2 elems each
  const float* x = X + (long)row * DIM;
  __shared__ float red[256];
  float a = x[tid], c = x[tid + 256];
  red[tid] = a + c;
  __syncthreads();
  for (int off = 128; off > 0; off >>= 1) {
    if (tid < off) red[tid] += red[tid + off];
    __syncthreads();
  }
  float mean = red[0] * (1.0f / DIM);
  __syncthreads();
  float d0 = a - mean, d1 = c - mean;
  red[tid] = d0 * d0 + d1 * d1;
  __syncthreads();
  for (int off = 128; off > 0; off >>= 1) {
    if (tid < off) red[tid] += red[tid + off];
    __syncthreads();
  }
  float rstd = rsqrtf(red[0] * (1.0f / DIM) + 1e-5f);
  long o = (long)row * DIM;
  out[o + tid]       = f2bf(d0 * rstd * g[tid]       + bt[tid]);
  out[o + tid + 256] = f2bf(d1 * rstd * g[tid + 256] + bt[tid + 256]);
}

// ---------------------------------------------------------------- WMMA GEMM (LDS-staged, double-buffered)
// C[M,N] = A[M,K](bf16) @ W[N,K]^T(bf16)  (+ bias, + residual / gelu->bf16)
// mode 0: Cf = acc     mode 1: Cf = resid + acc + bias     mode 2: Cbf = bf16(gelu(acc+bias))
// Block: 128 threads (4 waves) -> 64x64 tile; wave w -> rows [w*16, w*16+16) x 64 cols.
// Tiles staged global->LDS with GLOBAL_LOAD_ASYNC_TO_LDS_B128 (ASYNCcnt); the k+1
// tile's async DMA overlaps the k tile's WMMA burst (double buffer).
__global__ __launch_bounds__(128)
void k_gemm(const u16* __restrict__ A, const u16* __restrict__ Bw,
            const float* __restrict__ bias, const float* __restrict__ resid,
            float* __restrict__ Cf, u16* __restrict__ Cbf,
            int M, int N, int K, int mode) {
  __shared__ u16 tA[2][TILE_U16];
  __shared__ u16 tB[2][TILE_U16];
  int tid  = threadIdx.x;
  int wave = tid >> 5, lane = tid & 31;
  int n0 = blockIdx.x * BN;
  int m0 = blockIdx.y * BM;
  int lh = lane >> 4;
  int kA = lh * 8;                          // A: K-offset per lane half
  int kB = lh * 16;                         // B: K-half per lane half
  unsigned ldsA = (unsigned)(size_t)&tA[0][0];
  unsigned ldsB = (unsigned)(size_t)&tB[0][0];
  int srow = tid >> 3, sseg = tid & 7;      // staging: this thread's first segment
  auto stage = [&](int buf, int k0) {
    unsigned boff = (unsigned)buf * (TILE_U16 * 2);
#pragma unroll
    for (int i = 0; i < 4; ++i) {           // 512 16B segments per tile / 128 threads
      int row = srow + i * 16;              // tid+i*128 -> row advances by 16
      async_b128(ldsA + boff + row * 144 + sseg * 16,
                 A  + (size_t)(m0 + row) * K + k0 + sseg * 8);
      async_b128(ldsB + boff + row * 144 + sseg * 16,
                 Bw + (size_t)(n0 + row) * K + k0 + sseg * 8);
    }
  };
  v8f acc[4] = {{}, {}, {}, {}};
  stage(0, 0);
  int cur = 0;
  for (int k0 = 0; k0 < K; k0 += BK) {
    wait_async0();                          // this wave's async copies complete
    __syncthreads();                        // all tiles written; prev reads done
    if (k0 + BK < K) stage(cur ^ 1, k0 + BK);   // overlap next DMA with compute
#pragma unroll
    for (int ks = 0; ks < 2; ++ks) {        // two 16x16x32 k-substeps
      union { v16bf v; u32 u[8]; } a, bfr[4];
      const u16* arow = &tA[cur][(wave * 16 + (lane & 15)) * LDSW + ks * 32];
#pragma unroll
      for (int pi = 0; pi < 8; ++pi) {
        int kk = kA + (pi & 3) * 2 + (pi >> 2) * 16;      // A 16x32 bf16 layout
        a.u[pi] = *(const u32*)(arow + kk);
      }
#pragma unroll
      for (int j = 0; j < 4; ++j) {         // preload all B fragments first ...
        const u16* brow = &tB[cur][(j * 16 + (lane & 15)) * LDSW + ks * 32];
#pragma unroll
        for (int pi = 0; pi < 8; ++pi)
          bfr[j].u[pi] = *(const u32*)(brow + kA + (pi & 3) * 2 + (pi >> 2) * 16); // B 32x16 bf16 layout
      }
#pragma unroll
      for (int j = 0; j < 4; ++j)           // ... then issue WMMAs back-to-back
        acc[j] = __builtin_amdgcn_wmma_f32_16x16x32_bf16(false, a.v, false, bfr[j].v,
                                                         (short)0, acc[j], false, false);
    }
    cur ^= 1;
  }
  int mbase = m0 + wave * 16;
#pragma unroll
  for (int j = 0; j < 4; ++j) {
    int col = n0 + j * 16 + (lane & 15);
#pragma unroll
    for (int r = 0; r < 8; ++r) {
      int  mr  = mbase + r + 8 * lh;                      // C/D f32 layout
      size_t idx = (size_t)mr * N + col;
      float v = acc[j][r];
      if (mode == 0) {
        Cf[idx] = v;
      } else if (mode == 1) {
        Cf[idx] = resid[idx] + v + bias[col];
      } else {
        float x = v + bias[col];
        float gl = 0.5f * x * (1.0f + erff(x * 0.70710678118654752f));
        Cbf[idx] = f2bf(gl);
      }
    }
  }
}

// ---------------------------------------------------------------- LSH bucket + stable sort
__global__ void k_lsh_sort(const float* __restrict__ QK, const float* __restrict__ rot,
                           int* __restrict__ st, int* __restrict__ inv) {
  int bh = blockIdx.x, b = bh >> 3, h = bh & 7;
  __shared__ float rotS[DHD * NHASH * 16];    // [f][round][i]
  __shared__ int   bucketArr[TOK];
  __shared__ int   hist[NBUK];
  __shared__ int   pref[NBUK];
  int t = threadIdx.x;                        // 512 threads, one token each
  for (int i = t; i < DHD * NHASH * 16; i += blockDim.x) rotS[i] = rot[i];
  float q[DHD];
  const float* qp = QK + ((long)b * TOK + t) * DIM + h * DHD;
#pragma unroll
  for (int f = 0; f < DHD; ++f) q[f] = qp[f];
  __syncthreads();
  for (int r = 0; r < NHASH; ++r) {
    float best = -1e30f; int bi = 0;
    for (int i = 0; i < 16; ++i) {
      float s = 0.f;
      for (int f = 0; f < DHD; ++f) s += q[f] * rotS[f * 128 + r * 16 + i];
      if (s > best)  { best = s;  bi = i; }
      if (-s > best) { best = -s; bi = i + 16; }
    }
    bucketArr[t] = bi;
    if (t < NBUK) hist[t] = 0;
    __syncthreads();
    atomicAdd(&hist[bi], 1);
    __syncthreads();
    if (t == 0) { int a = 0; for (int i = 0; i < NBUK; ++i) { pref[i] = a; a += hist[i]; } }
    __syncthreads();
    if (t < NBUK) {                          // bucket t: append tokens in t-order (stable)
      int pos = pref[t];
      long base = (long)bh * NHASH * TOK;
      for (int tt = 0; tt < TOK; ++tt)
        if (bucketArr[tt] == t) {
          int slot = r * TOK + pos;
          st[base + slot] = tt;
          inv[base + r * TOK + tt] = slot;
          ++pos;
        }
    }
    __syncthreads();
  }
}

// ---------------------------------------------------------------- per-chunk LSE (pass 1)
__global__ void k_chunk_lse(const float* __restrict__ QK, const int* __restrict__ st,
                            float* __restrict__ logits) {
  int c = blockIdx.x, bh = blockIdx.y;
  int b = bh >> 3, h = bh & 7;
  int r = c >> 5;                               // 32 chunks per round
  int pc = (c + NCHUNK - 1) % NCHUNK;           // global lookback (matches jnp.roll)
  __shared__ float qS[16][DHD];
  __shared__ float kS[32][DHD];
  __shared__ float dotS[16][32];
  __shared__ int   qt[16], kt[32];
  const int* stb = st + (long)bh * NHASH * TOK;
  int tid = threadIdx.x;                        // 128
  if (tid < 16) qt[tid] = stb[c * 16 + tid];
  if (tid < 32) kt[tid] = (tid < 16) ? stb[c * 16 + tid] : stb[pc * 16 + (tid - 16)];
  __syncthreads();
  for (int i = tid; i < 16 * DHD; i += 128) {
    int row = i >> 6, f = i & 63;
    qS[row][f] = QK[((long)b * TOK + qt[row]) * DIM + h * DHD + f];
  }
  for (int i = tid; i < 32 * DHD; i += 128) {
    int row = i >> 6, f = i & 63;
    kS[row][f] = QK[((long)b * TOK + kt[row]) * DIM + h * DHD + f];
  }
  __syncthreads();
  if (tid < 32) {                               // normalize keys (shared-QK)
    float s = 0.f;
    for (int f = 0; f < DHD; ++f) { float v = kS[tid][f]; s += v * v; }
    float invn = 1.0f / fmaxf(sqrtf(s), 1e-12f);
    for (int f = 0; f < DHD; ++f) kS[tid][f] *= invn;
  }
  __syncthreads();
  for (int pidx = tid; pidx < 16 * 32; pidx += 128) {
    int i = pidx >> 5, j = pidx & 31;
    float s = 0.f;
    for (int f = 0; f < DHD; ++f) s += qS[i][f] * kS[j][f];
    s *= 0.125f;                                // DH^-0.5
    if (qt[i] == kt[j]) s = -5e4f;              // TOKEN_SELF_ATTN_VALUE
    dotS[i][j] = s;
  }
  __syncthreads();
  if (tid < 16) {
    float m = -1e30f;
    for (int j = 0; j < 32; ++j) m = fmaxf(m, dotS[tid][j]);
    float sum = 0.f;
    for (int j = 0; j < 32; ++j) sum += expf(dotS[tid][j] - m);
    logits[((long)bh * NHASH + r) * TOK + qt[tid]] = m + logf(sum);
  }
}

// ---------------------------------------------------------------- softmax over hash rounds
__global__ void k_round_softmax(const float* __restrict__ logits,
                                float* __restrict__ mB, float* __restrict__ sB) {
  int idx = blockIdx.x * blockDim.x + threadIdx.x;
  if (idx >= BHN * TOK) return;
  int bh = idx >> 9, t = idx & 511;
  float m = -1e30f;
  for (int r = 0; r < NHASH; ++r)
    m = fmaxf(m, logits[((long)bh * NHASH + r) * TOK + t]);
  float s = 0.f;
  for (int r = 0; r < NHASH; ++r)
    s += expf(logits[((long)bh * NHASH + r) * TOK + t] - m);
  mB[idx] = m; sB[idx] = s;
}

// ---------------------------------------------------------------- per-token output (pass 2, deterministic)
__global__ void k_token_out(const float* __restrict__ QK, const float* __restrict__ Vb,
                            const int* __restrict__ st, const int* __restrict__ inv,
                            const float* __restrict__ logits,
                            const float* __restrict__ mB, const float* __restrict__ sB,
                            float* __restrict__ Ob) {
  int t = blockIdx.x, bh = blockIdx.y;
  int b = bh >> 3, h = bh & 7;
  int tid = threadIdx.x;                        // 64 threads, one feature each
  __shared__ float qS[DHD];
  __shared__ float pS[32];
  __shared__ int   ktS[32];
  if (tid < DHD) qS[tid] = QK[((long)b * TOK + t) * DIM + h * DHD + tid];
  float acc = 0.f;
  int   widx = bh * TOK + t;
  float wm = mB[widx], wsum = sB[widx];
  const int* stb = st + (long)bh * NHASH * TOK;
  for (int r = 0; r < NHASH; ++r) {
    int s  = inv[(long)bh * NHASH * TOK + r * TOK + t];
    int c  = s >> 4;
    int pc = (c + NCHUNK - 1) % NCHUNK;
    __syncthreads();
    if (tid < 32)
      ktS[tid] = (tid < 16) ? stb[c * 16 + tid] : stb[pc * 16 + (tid - 16)];
    __syncthreads();
    float lse = logits[((long)bh * NHASH + r) * TOK + t];
    if (tid < 32) {
      const float* kp = QK + ((long)b * TOK + ktS[tid]) * DIM + h * DHD;
      float ss = 0.f, dd = 0.f;
      for (int f = 0; f < DHD; ++f) { float kv = kp[f]; ss += kv * kv; dd += kv * qS[f]; }
      float d = (dd / fmaxf(sqrtf(ss), 1e-12f)) * 0.125f;
      if (ktS[tid] == t) d = -5e4f;
      pS[tid] = expf(d - lse);
    }
    __syncthreads();
    float w = expf(lse - wm) / wsum;
    float o = 0.f;
    for (int j = 0; j < 32; ++j)
      o += pS[j] * Vb[((long)b * TOK + ktS[j]) * DIM + h * DHD + tid];
    acc += w * o;
  }
  Ob[((long)b * TOK + t) * DIM + h * DHD + tid] = acc;
}

// ---------------------------------------------------------------- final heads
__global__ void k_head(const float* __restrict__ X1, const float* __restrict__ X2,
                       const float* fc1w, const float* fc1b,
                       const float* w21, const float* b21,
                       const float* w22, const float* b22,
                       const float* w23, const float* b23,
                       const float* w3,  const float* b3,
                       const float* w4,  const float* b4,
                       const int* cs, const int* co, const int* ss, const int* so,
                       float* __restrict__ out) {
  int b = blockIdx.x, tid = threadIdx.x;        // 512 threads
  __shared__ float hid[DIM];
  __shared__ float pred[DIM];
  __shared__ float logit[32];
  __shared__ float supls[3];
  long base = ((long)b * TOK + (TOK - 1)) * DIM;
  hid[tid] = 0.5f * (X1[base + tid] + X2[base + tid]);
  __syncthreads();
  {
    const float* wr = fc1w + (long)tid * 846;
    float s = 0.f;
    for (int k = 0; k < DIM; ++k) s += hid[k] * wr[k];
    s += wr[512 + cs[b]] + wr[662 + co[b]] + wr[812 + ss[b]] + wr[829 + so[b]] + fc1b[tid];
    pred[tid] = fmaxf(s, 0.f);
  }
  __syncthreads();
  if (tid < 3) {
    const float* wr = w3 + tid * DIM;
    float s = 0.f; for (int k = 0; k < DIM; ++k) s += pred[k] * wr[k];
    logit[tid] = s + b3[tid];
  }
  __syncthreads();
  if (tid == 0) {
    float m = fmaxf(fmaxf(logit[0], logit[1]), logit[2]);
    float sum = expf(logit[0] - m) + expf(logit[1] - m) + expf(logit[2] - m);
    float lse = m + logf(sum);
    for (int k = 0; k < 3; ++k) { supls[k] = logit[k] - lse; out[1600 + b * 3 + k] = supls[k]; }
  }
  __syncthreads();
  if (tid < 15) {
    const float* wr = w21 + tid * DIM;
    float s = 0.f; for (int k = 0; k < DIM; ++k) s += pred[k] * wr[k];
    logit[tid] = s + b21[tid];
  }
  __syncthreads();
  if (tid == 0) {
    float m = -1e30f; for (int k = 0; k < 15; ++k) m = fmaxf(m, logit[k]);
    float sum = 0.f;  for (int k = 0; k < 15; ++k) sum += expf(logit[k] - m);
    float lse = m + logf(sum);
    for (int k = 0; k < 15; ++k) out[b * 15 + k] = logit[k] - lse + supls[0];
  }
  __syncthreads();
  if (tid < 11) {
    const float* wr = w22 + tid * DIM;
    float s = 0.f; for (int k = 0; k < DIM; ++k) s += pred[k] * wr[k];
    logit[tid] = s + b22[tid];
  }
  __syncthreads();
  if (tid == 0) {
    float m = -1e30f; for (int k = 0; k < 11; ++k) m = fmaxf(m, logit[k]);
    float sum = 0.f;  for (int k = 0; k < 11; ++k) sum += expf(logit[k] - m);
    float lse = m + logf(sum);
    for (int k = 0; k < 11; ++k) out[480 + b * 11 + k] = logit[k] - lse + supls[1];
  }
  __syncthreads();
  if (tid < 24) {
    const float* wr = w23 + tid * DIM;
    float s = 0.f; for (int k = 0; k < DIM; ++k) s += pred[k] * wr[k];
    logit[tid] = s + b23[tid];
  }
  __syncthreads();
  if (tid == 0) {
    float m = -1e30f; for (int k = 0; k < 24; ++k) m = fmaxf(m, logit[k]);
    float sum = 0.f;  for (int k = 0; k < 24; ++k) sum += expf(logit[k] - m);
    float lse = m + logf(sum);
    for (int k = 0; k < 24; ++k) out[832 + b * 24 + k] = logit[k] - lse + supls[2];
    float s = 0.f; for (int k = 0; k < DIM; ++k) s += pred[k] * w4[k];
    out[1696 + b] = s + b4[0];
  }
}

// ================================================================ host
extern "C" void kernel_launch(void* const* d_in, const int* in_sizes, int n_in,
                              void* d_out, int out_size, void* d_ws, size_t ws_size,
                              hipStream_t stream) {
  (void)in_sizes; (void)n_in; (void)out_size; (void)ws_size;
  auto F = [&](int i) { return (const float*)d_in[i]; };

  // ---- workspace carve (aligned 256B); total ~283 MB
  char* wp = (char*)d_ws;
  auto carve = [&](size_t bytes) -> void* {
    void* r = (void*)wp; wp += (bytes + 255) & ~(size_t)255; return r;
  };
  float* X1   = (float*)carve((size_t)MROWS * DIM * 4);
  float* X2   = (float*)carve((size_t)MROWS * DIM * 4);
  float* QK   = (float*)carve((size_t)MROWS * DIM * 4);
  float* V    = (float*)carve((size_t)MROWS * DIM * 4);
  float* Ob   = (float*)carve((size_t)MROWS * DIM * 4);
  u16*   XNbf = (u16*)carve((size_t)MROWS * DIM * 2);
  u16*   Hbf  = (u16*)carve((size_t)MROWS * FFD * 2);
  u16 *Wqkb[3], *Wvb[3], *Wob[3], *W1b[3], *W2b[3];
  for (int l = 0; l < 3; ++l) {
    Wqkb[l] = (u16*)carve((size_t)DIM * DIM * 2);
    Wvb[l]  = (u16*)carve((size_t)DIM * DIM * 2);
    Wob[l]  = (u16*)carve((size_t)DIM * DIM * 2);
    W1b[l]  = (u16*)carve((size_t)FFD * DIM * 2);
    W2b[l]  = (u16*)carve((size_t)DIM * FFD * 2);
  }
  int*   st   = (int*)carve((size_t)BHN * NHASH * TOK * 4);
  int*   inv  = (int*)carve((size_t)BHN * NHASH * TOK * 4);
  float* lg   = (float*)carve((size_t)BHN * NHASH * TOK * 4);
  float* mB   = (float*)carve((size_t)BHN * TOK * 4);
  float* sB   = (float*)carve((size_t)BHN * TOK * 4);

  // ---- convert all GEMM weights to bf16
  for (int l = 0; l < 3; ++l) {
    int base = l * 12;
    long nq = (long)DIM * DIM, nf = (long)FFD * DIM;
    k_convert<<<dim3((nq + 255) / 256), dim3(256), 0, stream>>>(F(base + 2),  Wqkb[l], nq);
    k_convert<<<dim3((nq + 255) / 256), dim3(256), 0, stream>>>(F(base + 3),  Wvb[l],  nq);
    k_convert<<<dim3((nq + 255) / 256), dim3(256), 0, stream>>>(F(base + 4),  Wob[l],  nq);
    k_convert<<<dim3((nf + 255) / 256), dim3(256), 0, stream>>>(F(base + 8),  W1b[l],  nf);
    k_convert<<<dim3((nf + 255) / 256), dim3(256), 0, stream>>>(F(base + 10), W2b[l],  nf);
  }

  // ---- organize (pool + PE + ones) -> X1 = X2
  long ntot = (long)BATCH * TOK * DIM;
  k_organize<<<dim3((ntot + 255) / 256), dim3(256), 0, stream>>>(F(49), F(50), X1, X2);

  // ---- reversible reformer layers
  const float* rot = F(48);
  for (int l = 0; l < 3; ++l) {
    int base = l * 12;
    dim3 gq(DIM / BN, MROWS / BM), gf(FFD / BN, MROWS / BM);
    // x1 += attn(LN(x2))
    k_ln<<<dim3(MROWS), dim3(256), 0, stream>>>(X2, F(base + 0), F(base + 1), XNbf);
    k_gemm<<<gq, dim3(128), 0, stream>>>(XNbf, Wqkb[l], nullptr, nullptr, QK, nullptr, MROWS, DIM, DIM, 0);
    k_gemm<<<gq, dim3(128), 0, stream>>>(XNbf, Wvb[l],  nullptr, nullptr, V,  nullptr, MROWS, DIM, DIM, 0);
    k_lsh_sort<<<dim3(BHN), dim3(512), 0, stream>>>(QK, rot + (size_t)l * DHD * NHASH * 16, st, inv);
    k_chunk_lse<<<dim3(NCHUNK, BHN), dim3(128), 0, stream>>>(QK, st, lg);
    k_round_softmax<<<dim3((BHN * TOK + 255) / 256), dim3(256), 0, stream>>>(lg, mB, sB);
    k_token_out<<<dim3(TOK, BHN), dim3(64), 0, stream>>>(QK, V, st, inv, lg, mB, sB, Ob);
    k_convert<<<dim3((int)((ntot + 255) / 256)), dim3(256), 0, stream>>>(Ob, XNbf, ntot);
    k_gemm<<<gq, dim3(128), 0, stream>>>(XNbf, Wob[l], F(base + 5), X1, X1, nullptr, MROWS, DIM, DIM, 1);
    // x2 += ff(LN(x1))
    k_ln<<<dim3(MROWS), dim3(256), 0, stream>>>(X1, F(base + 6), F(base + 7), XNbf);
    k_gemm<<<gf, dim3(128), 0, stream>>>(XNbf, W1b[l], F(base + 9), nullptr, nullptr, Hbf, MROWS, FFD, DIM, 2);
    k_gemm<<<gq, dim3(128), 0, stream>>>(Hbf, W2b[l], F(base + 11), X2, X2, nullptr, MROWS, DIM, FFD, 1);
  }

  // ---- heads
  k_head<<<dim3(BATCH), dim3(512), 0, stream>>>(
      X1, X2,
      F(36), F(37), F(38), F(39), F(40), F(41), F(42), F(43), F(44), F(45), F(46), F(47),
      (const int*)d_in[51], (const int*)d_in[52], (const int*)d_in[53], (const int*)d_in[54],
      (float*)d_out);
}
